// DGCNN_19378892440053
// MI455X (gfx1250) — matrix operand order repaired and checked by reading, not compiled, not run on hardware
//
#include <hip/hip_runtime.h>
#include <hip/hip_bf16.h>

typedef float    v8f  __attribute__((ext_vector_type(8)));
typedef _Float16 v8h  __attribute__((ext_vector_type(8)));
typedef _Float16 v16h __attribute__((ext_vector_type(16)));

#define BATCH 16
#define NPTS  2048
#define KNN   20

__device__ __forceinline__ float lrelu(float x) { return x > 0.f ? x : 0.2f * x; }

// order-preserving float<->uint encoding for atomicMax
__device__ __forceinline__ unsigned encF(float f) {
    unsigned u = __float_as_uint(f);
    return (u & 0x80000000u) ? ~u : (u | 0x80000000u);
}
__device__ __forceinline__ float decF(unsigned e) {
    return (e & 0x80000000u) ? __uint_as_float(e & 0x7FFFFFFFu) : __uint_as_float(~e);
}

// ---------------------------------------------------------------------------
// Scalar kNN (layer 1, d=3 only): Gram-via-WMMA cannot pay for K=3.
// ---------------------------------------------------------------------------
template <int D>
__global__ void knn_kernel(const float* __restrict__ x, int xstride,
                           int* __restrict__ idx) {
    constexpr int TILE = 128;
    __shared__ float xT[TILE][D];
    __shared__ float dl[TILE][KNN];
    __shared__ int   il[TILE][KNN];

    const int tid = threadIdx.x;
    const int blocksPerB = NPTS / TILE;
    const int b = blockIdx.x / blocksPerB;
    const int n = (blockIdx.x % blocksPerB) * TILE + tid;

    float xi[D];
#pragma unroll
    for (int c = 0; c < D; ++c) xi[c] = x[((size_t)b * NPTS + n) * xstride + c];
#pragma unroll
    for (int k = 0; k < KNN; ++k) { dl[tid][k] = __builtin_inff(); il[tid][k] = 0; }

    for (int jt = 0; jt < NPTS; jt += TILE) {
        __syncthreads();
        for (int e = tid; e < TILE * D; e += TILE) {
            int r = e / D, c = e % D;
            xT[r][c] = x[((size_t)b * NPTS + jt + r) * xstride + c];
        }
        __syncthreads();
        for (int jj = 0; jj < TILE; ++jj) {
            float d2 = 0.f;
#pragma unroll
            for (int c = 0; c < D; ++c) { float df = xi[c] - xT[jj][c]; d2 += df * df; }
            if (d2 < dl[tid][KNN - 1]) {
                int p = KNN - 1;
                while (p > 0 && dl[tid][p - 1] > d2) {
                    dl[tid][p] = dl[tid][p - 1]; il[tid][p] = il[tid][p - 1]; --p;
                }
                dl[tid][p] = d2; il[tid][p] = jt + jj;
            }
        }
    }
    __syncthreads();
    for (int k = 0; k < KNN; ++k) idx[((size_t)b * NPTS + n) * KNN + k] = il[tid][k];
}

// ---------------------------------------------------------------------------
// f16 feature copy + squared norms (norms from f16-rounded values so the
// WMMA Gram matrix is self-consistent and self-distance stays ~0).
// ---------------------------------------------------------------------------
__global__ void pack_xh(const float* __restrict__ x /* xc slice, stride 256 */,
                        _Float16* __restrict__ xh, float* __restrict__ sqn) {
    int i = blockIdx.x * blockDim.x + threadIdx.x;
    if (i >= BATCH * NPTS) return;
    const float* r = x + (size_t)i * 256;
    float s = 0.f;
#pragma unroll
    for (int c = 0; c < 64; ++c) {
        _Float16 h = (_Float16)r[c];
        xh[(size_t)i * 64 + c] = h;
        float f = (float)h;
        s += f * f;
    }
    sqn[i] = s;
}

// ---------------------------------------------------------------------------
// WMMA kNN (layers 2/3, d=64): block = 128 points (4 waves).
// Gram tile xi . xj^T via v_wmma f16 (A-frags register-resident across all j),
// d2 = |xi|^2 + |xj|^2 - 2 dot (reference's exact expansion), then per-thread
// top-20 insertion scan over the LDS dot tile (rows padded -> no bank clash).
// ---------------------------------------------------------------------------
__global__ void knn_wmma(const _Float16* __restrict__ xh,
                         const float* __restrict__ sqn,
                         int* __restrict__ idx) {
    constexpr int TI = 128, TJ = 64, D = 64;
    __shared__ float dotL[TI][TJ + 1];
    __shared__ float dl[TI][KNN];
    __shared__ int   il[TI][KNN];
    __shared__ float sqJ[TJ];

    const int tid = threadIdx.x;  // 128 threads
    const int blocksPerB = NPTS / TI;
    const int b = blockIdx.x / blocksPerB;
    const int i0 = (blockIdx.x % blocksPerB) * TI;
    const size_t base = (size_t)b * NPTS;

    const float sqi = sqn[base + i0 + tid];
#pragma unroll
    for (int k = 0; k < KNN; ++k) { dl[tid][k] = __builtin_inff(); il[tid][k] = 0; }

    const int wave = tid / 32, lane = tid % 32;
    const int half = lane / 16, l16 = lane % 16;

    // A-frags for this wave's two M-tiles: invariant over j, keep in VGPRs.
    v16h Af[2][2];
#pragma unroll
    for (int mt = 0; mt < 2; ++mt) {
        const int m = (2 * wave + mt) * 16 + l16;
        const _Float16* ar = xh + (base + i0 + m) * D;
#pragma unroll
        for (int ks = 0; ks < 2; ++ks) {
            v8h a0 = *(const v8h*)(ar + ks * 32 + 16 * half);
            v8h a1 = *(const v8h*)(ar + ks * 32 + 16 * half + 8);
#pragma unroll
            for (int q = 0; q < 8; ++q) { Af[mt][ks][q] = a0[q]; Af[mt][ks][q + 8] = a1[q]; }
        }
    }

    for (int jt = 0; jt < NPTS; jt += TJ) {
        // B-frags: lane = N (column = point j), K halves contiguous.
        v16h Bf[4][2];
#pragma unroll
        for (int nt = 0; nt < 4; ++nt) {
            const _Float16* br = xh + (base + jt + nt * 16 + l16) * D;
#pragma unroll
            for (int ks = 0; ks < 2; ++ks) {
                v8h b0 = *(const v8h*)(br + ks * 32 + 16 * half);
                v8h b1 = *(const v8h*)(br + ks * 32 + 16 * half + 8);
#pragma unroll
                for (int q = 0; q < 8; ++q) { Bf[nt][ks][q] = b0[q]; Bf[nt][ks][q + 8] = b1[q]; }
            }
        }
        v8f acc[2][4];
#pragma unroll
        for (int mt = 0; mt < 2; ++mt)
#pragma unroll
            for (int nt = 0; nt < 4; ++nt) {
                v8f z = {0.f, 0.f, 0.f, 0.f, 0.f, 0.f, 0.f, 0.f};
                acc[mt][nt] = z;
            }
        // 8 independent accumulator chains -> deep WMMA pipelining
#pragma unroll
        for (int ks = 0; ks < 2; ++ks)
#pragma unroll
            for (int mt = 0; mt < 2; ++mt)
#pragma unroll
                for (int nt = 0; nt < 4; ++nt)
                    acc[mt][nt] = __builtin_amdgcn_wmma_f32_16x16x32_f16(
                        false, Af[mt][ks], false, Bf[nt][ks], (short)0,
                        acc[mt][nt], false, false);
#pragma unroll
        for (int mt = 0; mt < 2; ++mt)
#pragma unroll
            for (int nt = 0; nt < 4; ++nt)
#pragma unroll
                for (int r = 0; r < 8; ++r)
                    dotL[(2 * wave + mt) * 16 + r + 8 * half][nt * 16 + l16] = acc[mt][nt][r];
        if (tid < TJ) sqJ[tid] = sqn[base + jt + tid];
        __syncthreads();

        // top-20 insertion scan over this j-tile
        for (int jj = 0; jj < TJ; ++jj) {
            float d2 = sqi + sqJ[jj] - 2.f * dotL[tid][jj];
            if (d2 < dl[tid][KNN - 1]) {
                int p = KNN - 1;
                while (p > 0 && dl[tid][p - 1] > d2) {
                    dl[tid][p] = dl[tid][p - 1]; il[tid][p] = il[tid][p - 1]; --p;
                }
                dl[tid][p] = d2; il[tid][p] = jt + jj;
            }
        }
        __syncthreads();
    }
    for (int k = 0; k < KNN; ++k) idx[(base + i0 + tid) * KNN + k] = il[tid][k];
}

// ---------------------------------------------------------------------------
// pack W [K2][C] f32 -> Wt [C][D2PAD] f16 (transposed, K zero-padded)
// ---------------------------------------------------------------------------
__global__ void pack_w(const float* __restrict__ W, _Float16* __restrict__ Wt,
                       int K2, int C, int D2PAD) {
    int i = blockIdx.x * blockDim.x + threadIdx.x;
    if (i >= C * D2PAD) return;
    int c = i / D2PAD, k = i % D2PAD;
    Wt[i] = (k < K2) ? (_Float16)W[(size_t)k * C + c] : (_Float16)0.f;
}

__global__ void zero_f32(float* p, int n) {
    int i = blockIdx.x * blockDim.x + threadIdx.x;
    if (i < n) p[i] = 0.f;
}
__global__ void zero_u32(unsigned* p, int n) {
    int i = blockIdx.x * blockDim.x + threadIdx.x;
    if (i < n) p[i] = 0u;
}

// ---------------------------------------------------------------------------
// EdgeConv core: 4 points x 20 neighbors = 80 edge rows per block.
// e = [x_i, x_j - x_i] f16, h = e @ W via v_wmma; fused BN-stat accumulation
// (sum/sumsq of pre-activation h) + raw max-over-k (BN+LeakyReLU applied
// later -- valid since BN scale > 0 and both are monotone).
// ---------------------------------------------------------------------------
template <int D, int D2PAD, int C>
__global__ void edge_gemm_max(const float* __restrict__ x, int xstride,
                              const int* __restrict__ idx,
                              const _Float16* __restrict__ Wt,   // [C][D2PAD]
                              float* __restrict__ xc, int coff,
                              float* __restrict__ gsum, float* __restrict__ gsumsq) {
    constexpr int PTS = 4, EDG = PTS * KNN;  // 80 = 5 * 16
    __shared__ __align__(16) _Float16 eL[EDG][D2PAD];
    __shared__ __align__(16) float    hL[EDG][C];

    const int tid = threadIdx.x;
    const int blocksPerB = NPTS / PTS;
    const int b = blockIdx.x / blocksPerB;
    const int n0 = (blockIdx.x % blocksPerB) * PTS;

    for (int e = tid; e < EDG * D2PAD; e += 256) {
        int row = e / D2PAD, col = e % D2PAD;
        int p = row / KNN, k = row % KNN;
        int n = n0 + p;
        float v = 0.f;
        if (col < D) {
            v = x[((size_t)b * NPTS + n) * xstride + col];
        } else if (col < 2 * D) {
            int c2 = col - D;
            int j = idx[((size_t)b * NPTS + n) * KNN + k];
            v = x[((size_t)b * NPTS + j) * xstride + c2] -
                x[((size_t)b * NPTS + n) * xstride + c2];
        }
        eL[row][col] = (_Float16)v;
    }
    __syncthreads();

    const int wave = tid / 32, lane = tid % 32;
    const int half = lane / 16, l16 = lane % 16;
    constexpr int MT = EDG / 16, NT = C / 16;

    for (int t = wave; t < MT * NT; t += 8) {
        const int mb = (t / NT) * 16, nb = (t % NT) * 16;
        v8f acc = {0.f, 0.f, 0.f, 0.f, 0.f, 0.f, 0.f, 0.f};
#pragma unroll
        for (int ks = 0; ks < D2PAD / 32; ++ks) {
            const _Float16* ar = &eL[mb + l16][ks * 32];
            v8h a0 = *(const v8h*)(ar + 16 * half);
            v8h a1 = *(const v8h*)(ar + 16 * half + 8);
            const _Float16* br = Wt + (size_t)(nb + l16) * D2PAD + ks * 32 + 16 * half;
            v8h b0 = *(const v8h*)(br);
            v8h b1 = *(const v8h*)(br + 8);
            v16h A, Bm;
#pragma unroll
            for (int i = 0; i < 8; ++i) {
                A[i] = a0[i]; A[i + 8] = a1[i];
                Bm[i] = b0[i]; Bm[i + 8] = b1[i];
            }
            acc = __builtin_amdgcn_wmma_f32_16x16x32_f16(false, A, false, Bm,
                                                         (short)0, acc, false, false);
        }
#pragma unroll
        for (int r = 0; r < 8; ++r)
            hL[mb + r + 8 * half][nb + l16] = acc[r];
    }
    __syncthreads();

    for (int c = tid; c < C; c += 256) {
        float s = 0.f, s2 = 0.f;
        for (int r = 0; r < EDG; ++r) { float v = hL[r][c]; s += v; s2 += v * v; }
        atomicAdd(&gsum[c], s);
        atomicAdd(&gsumsq[c], s2);
    }
    for (int pc = tid; pc < PTS * C; pc += 256) {
        int p = pc / C, c = pc % C;
        float m = -__builtin_inff();
        for (int k = 0; k < KNN; ++k) m = fmaxf(m, hL[p * KNN + k][c]);
        xc[((size_t)b * NPTS + n0 + p) * 256 + coff + c] = m;
    }
}

// apply BN(batch stats) + LeakyReLU in place on xc channel slice
__global__ void bn_edge_apply(float* __restrict__ xc, int coff, int C,
                              const float* __restrict__ gsum,
                              const float* __restrict__ gsumsq,
                              const float* __restrict__ gamma,
                              const float* __restrict__ beta, float cnt) {
    int i = blockIdx.x * blockDim.x + threadIdx.x;
    if (i >= BATCH * NPTS * C) return;
    int c = i % C;
    size_t pn = (size_t)i / C;
    float m = gsum[c] / cnt;
    float v = gsumsq[c] / cnt - m * m;
    float sc = gamma[c] * rsqrtf(v + 1e-5f);
    float val = xc[pn * 256 + coff + c];
    xc[pn * 256 + coff + c] = lrelu(sc * (val - m) + beta[c]);
}

// ---------------------------------------------------------------------------
// conv_final: h = xc[32768,256] @ Wf[256,1024] via WMMA; per-channel
// sum/sumsq over (B,N) + per-(batch,channel) max over N via encoded atomicMax.
// ---------------------------------------------------------------------------
__global__ void convf_kernel(const float* __restrict__ xc,
                             const _Float16* __restrict__ Wtf,  // [1024][256]
                             float* __restrict__ gsum, float* __restrict__ gsumsq,
                             unsigned* __restrict__ gmaxE) {
    constexpr int MR = 32, NC = 128, KD = 256;
    __shared__ __align__(16) _Float16 xL[MR][KD];
    __shared__ __align__(16) float    hL[MR][NC];

    const int tid = threadIdx.x;
    const size_t row0 = (size_t)blockIdx.x * MR;
    const int col0 = blockIdx.y * NC;

    for (int e = tid; e < MR * KD; e += 256) {
        int r = e / KD, c = e % KD;
        xL[r][c] = (_Float16)xc[(row0 + r) * KD + c];
    }
    __syncthreads();

    const int wave = tid / 32, lane = tid % 32;
    const int half = lane / 16, l16 = lane % 16;
    constexpr int MT = MR / 16, NT = NC / 16;  // 16 tiles, 8 waves

    for (int t = wave; t < MT * NT; t += 8) {
        const int mb = (t / NT) * 16, nb = (t % NT) * 16;
        v8f acc = {0.f, 0.f, 0.f, 0.f, 0.f, 0.f, 0.f, 0.f};
#pragma unroll
        for (int ks = 0; ks < KD / 32; ++ks) {
            const _Float16* ar = &xL[mb + l16][ks * 32];
            v8h a0 = *(const v8h*)(ar + 16 * half);
            v8h a1 = *(const v8h*)(ar + 16 * half + 8);
            const _Float16* br = Wtf + (size_t)(col0 + nb + l16) * KD + ks * 32 + 16 * half;
            v8h b0 = *(const v8h*)(br);
            v8h b1 = *(const v8h*)(br + 8);
            v16h A, Bm;
#pragma unroll
            for (int i = 0; i < 8; ++i) {
                A[i] = a0[i]; A[i + 8] = a1[i];
                Bm[i] = b0[i]; Bm[i + 8] = b1[i];
            }
            acc = __builtin_amdgcn_wmma_f32_16x16x32_f16(false, A, false, Bm,
                                                         (short)0, acc, false, false);
        }
#pragma unroll
        for (int r = 0; r < 8; ++r)
            hL[mb + r + 8 * half][nb + l16] = acc[r];
    }
    __syncthreads();

    if (tid < NC) {
        const int c = tid, cg = col0 + c;
        const int b = (int)(row0 / NPTS);
        float m = -__builtin_inff(), s = 0.f, s2 = 0.f;
        for (int r = 0; r < MR; ++r) {
            float v = hL[r][c];
            m = fmaxf(m, v); s += v; s2 += v * v;
        }
        atomicAdd(&gsum[cg], s);
        atomicAdd(&gsumsq[cg], s2);
        atomicMax(&gmaxE[b * 1024 + cg], encF(m));
    }
}

__global__ void bn_gfeat(const unsigned* __restrict__ gmaxE,
                         const float* __restrict__ gsum, const float* __restrict__ gsumsq,
                         const float* __restrict__ gamma, const float* __restrict__ beta,
                         float* __restrict__ gfeat) {
    int i = blockIdx.x * blockDim.x + threadIdx.x;
    if (i >= BATCH * 1024) return;
    int c = i % 1024;
    const float cnt = (float)(BATCH * NPTS);
    float m = gsum[c] / cnt;
    float v = gsumsq[c] / cnt - m * m;
    float val = decF(gmaxE[i]);
    gfeat[i] = lrelu(gamma[c] * (val - m) * rsqrtf(v + 1e-5f) + beta[c]);
}

// ---------------------------------------------------------------------------
// Classifier head (tiny): plain FC, BN over batch of 16, LeakyReLU.
// ---------------------------------------------------------------------------
__global__ void fc_kernel(const float* __restrict__ in, const float* __restrict__ W,
                          const float* __restrict__ bias, float* __restrict__ out,
                          int Kin, int Kout) {
    int i = blockIdx.x * blockDim.x + threadIdx.x;
    if (i >= BATCH * Kout) return;
    int b = i / Kout, o = i % Kout;
    float s = bias[o];
    for (int k = 0; k < Kin; ++k) s += in[b * Kin + k] * W[(size_t)k * Kout + o];
    out[i] = s;
}

__global__ void bn_batch_lrelu(float* __restrict__ h, const float* __restrict__ g,
                               const float* __restrict__ be, int Kout) {
    int o = blockIdx.x * blockDim.x + threadIdx.x;
    if (o >= Kout) return;
    float s = 0.f, s2 = 0.f;
    for (int b = 0; b < BATCH; ++b) { float v = h[b * Kout + o]; s += v; s2 += v * v; }
    float m = s / (float)BATCH;
    float v = s2 / (float)BATCH - m * m;
    float sc = g[o] * rsqrtf(v + 1e-5f);
    for (int b = 0; b < BATCH; ++b)
        h[b * Kout + o] = lrelu(sc * (h[b * Kout + o] - m) + be[o]);
}

// ---------------------------------------------------------------------------
extern "C" void kernel_launch(void* const* d_in, const int* in_sizes, int n_in,
                              void* d_out, int out_size, void* d_ws, size_t ws_size,
                              hipStream_t stream) {
    const float* pos = (const float*)d_in[0];
    const float* W1 = (const float*)d_in[1];  const float* g1 = (const float*)d_in[2];  const float* b1 = (const float*)d_in[3];
    const float* W2 = (const float*)d_in[4];  const float* g2 = (const float*)d_in[5];  const float* b2 = (const float*)d_in[6];
    const float* W3 = (const float*)d_in[7];  const float* g3 = (const float*)d_in[8];  const float* b3 = (const float*)d_in[9];
    const float* Wf = (const float*)d_in[10]; const float* gf = (const float*)d_in[11]; const float* bf = (const float*)d_in[12];
    const float* W4 = (const float*)d_in[13]; const float* b4 = (const float*)d_in[14];
    const float* g4 = (const float*)d_in[15]; const float* be4 = (const float*)d_in[16];
    const float* W5 = (const float*)d_in[17]; const float* b5 = (const float*)d_in[18];
    const float* g5 = (const float*)d_in[19]; const float* be5 = (const float*)d_in[20];
    const float* W6 = (const float*)d_in[21]; const float* b6 = (const float*)d_in[22];
    float* out = (float*)d_out;

    // workspace carve-up (~40 MB)
    char* ws = (char*)d_ws;
    int*      idxB = (int*)ws;                                   // 2,621,440
    float*    xc   = (float*)(ws + 2621440);                     // 33,554,432
    _Float16* Wt   = (_Float16*)(ws + 2621440 + 33554432);       // 524,288
    float*    gsum   = (float*)((char*)Wt + 524288);             // 1024 f32
    float*    gsumsq = gsum + 1024;                              // 1024 f32
    unsigned* gmaxE  = (unsigned*)(gsum + 2048);                 // 16*1024 u32
    float*    gfeat  = (float*)(gmaxE + 16384);                  // 16*1024 f32
    float*    h4     = gfeat + 16384;                            // 16*512
    float*    h5     = h4 + 8192;                                // 16*256
    _Float16* xh     = (_Float16*)(h5 + 4096);                   // 16*2048*64 f16 = 4 MB
    float*    sqn    = (float*)((char*)xh + 4194304);            // 16*2048 f32

    const float EDGE_CNT = (float)(BATCH * NPTS * KNN);
    const int knnBlocks = BATCH * (NPTS / 128);
    const int edgeBlocks = BATCH * (NPTS / 4);
    const int ptBlocks = (BATCH * NPTS + 255) / 256;

    // -------- EdgeConv 1: pos(d=3) -> xc[:,0:64] --------
    knn_kernel<3><<<knnBlocks, 128, 0, stream>>>(pos, 3, idxB);
    pack_w<<<(64 * 32 + 255) / 256, 256, 0, stream>>>(W1, Wt, 6, 64, 32);
    zero_f32<<<8, 256, 0, stream>>>(gsum, 2048);
    edge_gemm_max<3, 32, 64><<<edgeBlocks, 256, 0, stream>>>(pos, 3, idxB, Wt, xc, 0, gsum, gsumsq);
    bn_edge_apply<<<(BATCH * NPTS * 64 + 255) / 256, 256, 0, stream>>>(xc, 0, 64, gsum, gsumsq, g1, b1, EDGE_CNT);

    // -------- EdgeConv 2: x1(d=64) -> xc[:,64:128] --------
    pack_xh<<<ptBlocks, 256, 0, stream>>>(xc, xh, sqn);
    knn_wmma<<<knnBlocks, 128, 0, stream>>>(xh, sqn, idxB);
    pack_w<<<(64 * 128 + 255) / 256, 256, 0, stream>>>(W2, Wt, 128, 64, 128);
    zero_f32<<<8, 256, 0, stream>>>(gsum, 2048);
    edge_gemm_max<64, 128, 64><<<edgeBlocks, 256, 0, stream>>>(xc, 256, idxB, Wt, xc, 64, gsum, gsumsq);
    bn_edge_apply<<<(BATCH * NPTS * 64 + 255) / 256, 256, 0, stream>>>(xc, 64, 64, gsum, gsumsq, g2, b2, EDGE_CNT);

    // -------- EdgeConv 3: x2(d=64) -> xc[:,128:256] --------
    pack_xh<<<ptBlocks, 256, 0, stream>>>(xc + 64, xh, sqn);
    knn_wmma<<<knnBlocks, 128, 0, stream>>>(xh, sqn, idxB);
    pack_w<<<(128 * 128 + 255) / 256, 256, 0, stream>>>(W3, Wt, 128, 128, 128);
    zero_f32<<<8, 256, 0, stream>>>(gsum, 2048);
    edge_gemm_max<64, 128, 128><<<edgeBlocks, 256, 0, stream>>>(xc + 64, 256, idxB, Wt, xc, 128, gsum, gsumsq);
    bn_edge_apply<<<(BATCH * NPTS * 128 + 255) / 256, 256, 0, stream>>>(xc, 128, 128, gsum, gsumsq, g3, b3, EDGE_CNT);

    // -------- conv_final + global max pool --------
    pack_w<<<(1024 * 256 + 255) / 256, 256, 0, stream>>>(Wf, Wt, 256, 1024, 256);
    zero_f32<<<8, 256, 0, stream>>>(gsum, 2048);
    zero_u32<<<64, 256, 0, stream>>>(gmaxE, 16384);
    {
        dim3 grid(BATCH * NPTS / 32, 1024 / 128);
        convf_kernel<<<grid, 256, 0, stream>>>(xc, Wt, gsum, gsumsq, gmaxE);
    }
    bn_gfeat<<<(BATCH * 1024 + 255) / 256, 256, 0, stream>>>(gmaxE, gsum, gsumsq, gf, bf, gfeat);

    // -------- classifier head --------
    fc_kernel<<<(BATCH * 512 + 255) / 256, 256, 0, stream>>>(gfeat, W4, b4, h4, 1024, 512);
    bn_batch_lrelu<<<2, 256, 0, stream>>>(h4, g4, be4, 512);
    fc_kernel<<<(BATCH * 256 + 255) / 256, 256, 0, stream>>>(h4, W5, b5, h5, 512, 256);
    bn_batch_lrelu<<<1, 256, 0, stream>>>(h5, g5, be5, 256);
    fc_kernel<<<(BATCH * 40 + 255) / 256, 256, 0, stream>>>(h5, W6, b6, out, 256, 40);

    (void)in_sizes; (void)n_in; (void)out_size; (void)ws_size;
}
